// Attention_7284264534326
// MI455X (gfx1250) — matrix operand order repaired and checked by reading, not compiled, not run on hardware
//
#include <hip/hip_runtime.h>

// ---------------------------------------------------------------------------
// Attention block for MI455X (gfx1250, wave32, WMMA).
//   q = (x1 @ wq) * D^-0.5 ; k = x2 @ wk ; v = x2 @ wv
//   a = softmax(q @ k^T + mask * -1e9) ; y = relu(a @ v) @ wo
// All heavy contractions run on v_wmma_f32_16x16x32_bf16 (bf16 in, f32 acc).
// B-panels (<=4MB) live in the 192MB L2, so GEMM fragments load directly from
// global memory; no LDS staging needed for this shape.
// ---------------------------------------------------------------------------

static constexpr float MIN_NUMBER = -1.0e9f;

typedef __attribute__((ext_vector_type(16))) __bf16 v16bf;
typedef __attribute__((ext_vector_type(8)))  __bf16 v8bf;
typedef __attribute__((ext_vector_type(8)))  float  v8f;

union BFrag { v16bf v; v8bf h[2]; };

enum { EPI_F32 = 0, EPI_BF16 = 1, EPI_BF16_RELU = 2, EPI_BF16_T = 3 };

// A-operand fragment, 16x32 bf16, row-major source (ld = K).
// ISA layout: lanes 0-15 (M=0..15) hold K = 0..7 (v0-3) and 16..23 (v4-7);
// lanes 16-31 hold K = 8..15 and 24..31.
__device__ __forceinline__ v16bf load_frag_a(const __bf16* base, int ld,
                                             int row0, int k0, int lane) {
  const int hl = lane & 15, sel = lane >> 4;
  const __bf16* p = base + (size_t)(row0 + hl) * ld + k0 + sel * 8;
  BFrag u;
  u.h[0] = *(const v8bf*)(p);
  u.h[1] = *(const v8bf*)(p + 16);
  return u.v;
}

// B-operand fragment, 32x16 bf16, from Bt stored N x K row-major (ld = K).
// Layout matches A: lanes 0-15 (N=0..15) hold K=0..7 and 16..23,
// lanes 16-31 hold K=8..15 and 24..31.
__device__ __forceinline__ v16bf load_frag_b(const __bf16* base, int ld,
                                             int col0, int k0, int lane) {
  const int hl = lane & 15, sel = lane >> 4;
  const __bf16* p = base + (size_t)(col0 + hl) * ld + k0 + sel * 8;
  BFrag u;
  u.h[0] = *(const v8bf*)(p);
  u.h[1] = *(const v8bf*)(p + 16);
  return u.v;
}

// Generic batched GEMM: C[b] = A[b] (MxK, row-major) * Bt[b]^T (Bt is NxK).
// Wave computes 64x64 (4x4 WMMA tiles); block = 8 waves -> 128(M) x 256(N).
// Grid: x = N/256, y = M/128, z = batch.
__global__ __launch_bounds__(256, 1)
void gemm_wmma_bf16(const __bf16* __restrict__ A, const __bf16* __restrict__ Bt,
                    void* __restrict__ Cout, int K, int ldc,
                    size_t strideA, size_t strideB, size_t strideC,
                    float scale, int epi, const int* __restrict__ mask, int maskld)
{
  const int lane = threadIdx.x & 31;
  const int wave = threadIdx.x >> 5;
  const int b = blockIdx.z;
  const __bf16* Ab = A + (size_t)b * strideA;
  const __bf16* Bb = Bt + (size_t)b * strideB;
  const int row0 = blockIdx.y * 128 + (wave >> 2) * 64;
  const int col0 = blockIdx.x * 256 + (wave & 3) * 64;

  v8f acc[4][4] = {};

  for (int k0 = 0; k0 < K; k0 += 32) {
    v16bf af[4], bfv[4];
#pragma unroll
    for (int i = 0; i < 4; ++i) af[i] = load_frag_a(Ab, K, row0 + 16 * i, k0, lane);
#pragma unroll
    for (int j = 0; j < 4; ++j) bfv[j] = load_frag_b(Bb, K, col0 + 16 * j, k0, lane);

    if (k0 + 32 < K) {  // hint next K-slab into cache (global_prefetch_b8)
      __builtin_prefetch(Ab + (size_t)(row0 + (lane & 15)) * K + k0 + 32, 0, 3);
      __builtin_prefetch(Bb + (size_t)(col0 + (lane & 15)) * K + k0 + 32, 0, 3);
    }

#pragma unroll
    for (int i = 0; i < 4; ++i)
#pragma unroll
      for (int j = 0; j < 4; ++j)
        acc[i][j] = __builtin_amdgcn_wmma_f32_16x16x32_bf16(
            false, af[i], false, bfv[j], (short)0, acc[i][j], false, false);
  }

  // Epilogue. C-layout: VGPR t of lane -> row = t + (lane>>4)*8, col = lane&15.
  const int hl = lane & 15, sel = lane >> 4;
#pragma unroll
  for (int i = 0; i < 4; ++i) {
#pragma unroll
    for (int j = 0; j < 4; ++j) {
      const int rbase = row0 + 16 * i + sel * 8;
      const int c = col0 + 16 * j + hl;
      if (epi == EPI_F32) {
        const float madd =
            mask ? (float)mask[(size_t)b * maskld + c] * MIN_NUMBER : 0.0f;
        float* C = (float*)Cout + (size_t)b * strideC;
#pragma unroll
        for (int t = 0; t < 8; ++t)
          C[(size_t)(rbase + t) * ldc + c] = acc[i][j][t] * scale + madd;
      } else if (epi == EPI_BF16_T) {  // transposed bf16 store (for V^T)
        __bf16* C = (__bf16*)Cout + (size_t)b * strideC;
#pragma unroll
        for (int t = 0; t < 8; ++t)
          C[(size_t)c * ldc + (rbase + t)] = (__bf16)(acc[i][j][t]);
      } else {  // bf16 [+ relu], * scale
        __bf16* C = (__bf16*)Cout + (size_t)b * strideC;
#pragma unroll
        for (int t = 0; t < 8; ++t) {
          float v = acc[i][j][t] * scale;
          if (epi == EPI_BF16_RELU) v = fmaxf(v, 0.0f);
          C[(size_t)(rbase + t) * ldc + c] = (__bf16)v;
        }
      }
    }
  }
}

// ---------------- elementwise helpers ----------------

__global__ __launch_bounds__(256)
void f32_to_bf16(const float* __restrict__ in, __bf16* __restrict__ out, size_t n) {
  size_t i = (size_t)blockIdx.x * blockDim.x + threadIdx.x;
  const size_t stride = (size_t)gridDim.x * blockDim.x;
  for (; i < n; i += stride) out[i] = (__bf16)in[i];
}

// w (DxE, row-major, f32) -> wT (ExD, row-major, bf16)
__global__ __launch_bounds__(256)
void transpose_w(const float* __restrict__ w, __bf16* __restrict__ wT,
                 int D, int E) {
  const int idx = blockIdx.x * blockDim.x + threadIdx.x;
  if (idx >= D * E) return;
  const int d = idx / E, e = idx % E;
  wT[(size_t)e * D + d] = (__bf16)w[idx];
}

__device__ __forceinline__ float wredmax(float v) {
#pragma unroll
  for (int o = 16; o > 0; o >>= 1) v = fmaxf(v, __shfl_xor(v, o, 32));
  return v;
}
__device__ __forceinline__ float wredsum(float v) {
#pragma unroll
  for (int o = 16; o > 0; o >>= 1) v += __shfl_xor(v, o, 32);
  return v;
}

// One block per score row (S = 2048 = 256 threads * 8). f32 in, bf16 probs out.
__global__ __launch_bounds__(256)
void softmax_bf16(const float* __restrict__ Sc, __bf16* __restrict__ P, int S) {
  __shared__ float red[8];
  const size_t row = blockIdx.x;
  const float* x = Sc + row * (size_t)S;
  __bf16* p = P + row * (size_t)S;
  float v[8];
  float m = -3.0e38f;
#pragma unroll
  for (int i = 0; i < 8; ++i) {
    v[i] = x[threadIdx.x + 256 * i];
    m = fmaxf(m, v[i]);
  }
  m = wredmax(m);
  const int ln = threadIdx.x & 31, wid = threadIdx.x >> 5;
  if (ln == 0) red[wid] = m;
  __syncthreads();
  float mm = red[0];
#pragma unroll
  for (int k = 1; k < 8; ++k) mm = fmaxf(mm, red[k]);
  __syncthreads();
  float s = 0.f;
#pragma unroll
  for (int i = 0; i < 8; ++i) { v[i] = __expf(v[i] - mm); s += v[i]; }
  s = wredsum(s);
  if (ln == 0) red[wid] = s;
  __syncthreads();
  float tot = 0.f;
#pragma unroll
  for (int k = 0; k < 8; ++k) tot += red[k];
  const float inv = 1.0f / tot;
#pragma unroll
  for (int i = 0; i < 8; ++i)
    p[threadIdx.x + 256 * i] = (__bf16)(v[i] * inv);
}

// ---------------- host orchestration ----------------

extern "C" void kernel_launch(void* const* d_in, const int* in_sizes, int n_in,
                              void* d_out, int out_size, void* d_ws, size_t ws_size,
                              hipStream_t stream) {
  (void)in_sizes; (void)n_in; (void)out_size; (void)ws_size;
  constexpr int B = 8, S = 2048, D = 1024;
  constexpr size_t M = (size_t)B * S;  // 16384

  const float* x1 = (const float*)d_in[0];
  const float* x2 = (const float*)d_in[1];
  const int*   maskSeq = (const int*)d_in[2];
  const float* wq = (const float*)d_in[3];
  const float* wk = (const float*)d_in[4];
  const float* wv = (const float*)d_in[5];
  const float* wo = (const float*)d_in[6];

  char* ws = (char*)d_ws;
  size_t off = 0;
  auto take = [&](size_t bytes) -> void* {
    off = (off + 255) & ~(size_t)255;
    void* p = ws + off;
    off += bytes;
    return p;
  };

  __bf16* wqT = (__bf16*)take((size_t)D * D * 2);
  __bf16* wkT = (__bf16*)take((size_t)D * D * 2);
  __bf16* wvT = (__bf16*)take((size_t)D * D * 2);
  __bf16* woT = (__bf16*)take((size_t)D * D * 2);
  __bf16* Qb  = (__bf16*)take(M * D * 2);              // 32 MB
  __bf16* Kb  = (__bf16*)take(M * D * 2);              // 32 MB
  __bf16* Vt  = (__bf16*)take((size_t)B * D * S * 2);  // 32 MB, [b][e][s]
  __bf16* Yb  = (__bf16*)take(M * D * 2);              // 32 MB
  __bf16* P   = (__bf16*)take((size_t)B * S * S * 2);  // 64 MB
  __bf16* xb1 = (__bf16*)take(M * D * 2);              // 32 MB
  __bf16* xb2 = (__bf16*)take(M * D * 2);              // 32 MB
  // f32 scores (128 MB) alias xb1/xb2: both are dead before scores are made.
  // Peak ws use: ~328 MB.
  float* Sc = (float*)xb1;

  const dim3 blk(256);

  // 1) precision converts
  f32_to_bf16<<<8192, blk, 0, stream>>>(x1, xb1, M * D);
  f32_to_bf16<<<8192, blk, 0, stream>>>(x2, xb2, M * D);
  transpose_w<<<(D * D + 255) / 256, blk, 0, stream>>>(wq, wqT, D, D);
  transpose_w<<<(D * D + 255) / 256, blk, 0, stream>>>(wk, wkT, D, D);
  transpose_w<<<(D * D + 255) / 256, blk, 0, stream>>>(wv, wvT, D, D);
  transpose_w<<<(D * D + 255) / 256, blk, 0, stream>>>(wo, woT, D, D);

  // 2) Q = (x1 @ wq) * D^-0.5  (bf16 out)
  gemm_wmma_bf16<<<dim3(D / 256, M / 128, 1), blk, 0, stream>>>(
      xb1, wqT, Qb, D, D, 0, 0, 0, 0.03125f, EPI_BF16, nullptr, 0);
  // 3) K = x2 @ wk
  gemm_wmma_bf16<<<dim3(D / 256, M / 128, 1), blk, 0, stream>>>(
      xb2, wkT, Kb, D, D, 0, 0, 0, 1.0f, EPI_BF16, nullptr, 0);
  // 4) V = x2 @ wv, stored transposed per batch: Vt[b][e][s]
  gemm_wmma_bf16<<<dim3(D / 256, S / 128, B), blk, 0, stream>>>(
      xb2, wvT, Vt, D, /*ldc=*/S, (size_t)S * D, 0, (size_t)D * S,
      1.0f, EPI_BF16_T, nullptr, 0);

  // 5) scores = Q @ K^T + mask * -1e9  (f32 out, per batch)
  gemm_wmma_bf16<<<dim3(S / 256, S / 128, B), blk, 0, stream>>>(
      Qb, Kb, Sc, D, /*ldc=*/S, (size_t)S * D, (size_t)S * D, (size_t)S * S,
      1.0f, EPI_F32, maskSeq, S);

  // 6) softmax rows -> P (bf16)
  softmax_bf16<<<B * S, blk, 0, stream>>>(Sc, P, S);

  // 7) Y = relu(P @ V)  (Bt operand = Vt[b], N=E rows of length K=S)
  gemm_wmma_bf16<<<dim3(D / 256, S / 128, B), blk, 0, stream>>>(
      P, Vt, Yb, /*K=*/S, /*ldc=*/D, (size_t)S * S, (size_t)D * S,
      (size_t)S * D, 1.0f, EPI_BF16_RELU, nullptr, 0);

  // 8) out = Y @ wo  (f32 to d_out)
  gemm_wmma_bf16<<<dim3(D / 256, M / 128, 1), blk, 0, stream>>>(
      Yb, woT, d_out, D, D, 0, 0, 0, 1.0f, EPI_F32, nullptr, 0);
}
